// GNN_12652973654090
// MI455X (gfx1250) — matrix operand order repaired and checked by reading, not compiled, not run on hardware
//
#include <hip/hip_runtime.h>
#include <hip/hip_bf16.h>

typedef __attribute__((ext_vector_type(16))) _Float16 v16h;
typedef __attribute__((ext_vector_type(8)))  float    v8f;

#define HDIM 64
#define H2   128

__device__ __forceinline__ int a_kmap(int i, int hi) {
    // A-matrix 16x32 f16 fragment: half element i (0..15) -> K index
    // K = hi*16 + i
    return hi * 16 + i;
}

__device__ __forceinline__ v8f wmma_f16(v16h a, v16h b, v8f c) {
    return __builtin_amdgcn_wmma_f32_16x16x32_f16(
        /*neg_a=*/false, a, /*neg_b=*/false, b,
        /*c_mod=*/(short)0, c, /*reuse_a=*/false, /*reuse_b=*/false);
}

// ---------------------------------------------------------------------------
// Zero-fill utility (graph-capture safe; no hipMemset)
// ---------------------------------------------------------------------------
__global__ void k_zero(float* __restrict__ p, int n) {
    int i = blockIdx.x * blockDim.x + threadIdx.x;
    int stride = gridDim.x * blockDim.x;
    for (; i < n; i += stride) p[i] = 0.0f;
}

// ---------------------------------------------------------------------------
// Node embedding: h = x @ node_W + node_b   [N,32]@[32,64]
// One wave per 16-row tile. K=32 -> single WMMA per 16x16 output tile.
// Weights staged in LDS pre-swizzled to B-fragment order (2x ds_load_b128).
// ---------------------------------------------------------------------------
__global__ void k_node_embed(const float* __restrict__ x,
                             const float* __restrict__ W,
                             const float* __restrict__ b,
                             float* __restrict__ h, int N) {
    __shared__ __align__(32) _Float16 sW[4 * 32 * 16];  // [ct][lane][i]
    __shared__ float sb[HDIM];
    int t = threadIdx.x;
    for (int idx = t; idx < 32 * HDIM; idx += blockDim.x) {
        int k = idx >> 6, n = idx & 63;          // W[k][n], K=32
        int hi = (k >> 4) & 1, i = k & 15;       // B-frag: K = hi*16 + i
        int ct = n >> 4, ln = n & 15;
        sW[(ct * 32 + hi * 16 + ln) * 16 + i] = (_Float16)W[idx];
    }
    if (t < HDIM) sb[t] = b[t];
    __syncthreads();

    int wave = t >> 5, lane = t & 31, hi = lane >> 4, ln = lane & 15;
    int rowbase = (blockIdx.x * (blockDim.x >> 5) + wave) * 16;
    int arow = rowbase + ln;
    if (arow >= N) arow = N - 1;     // clamp: keeps EXEC all-1s for WMMA
    bool full = (rowbase + 16 <= N); // wave-uniform

    const float* xr = &x[arow * 32];
    v16h a;
#pragma unroll
    for (int i = 0; i < 16; ++i)
        a[i] = (_Float16)xr[a_kmap(i, hi)];

#pragma unroll
    for (int ct = 0; ct < 4; ++ct) {
        v16h bf = *(const v16h*)&sW[(ct * 32 + lane) * 16];
        v8f c;
        float bias = sb[ct * 16 + ln];
#pragma unroll
        for (int r = 0; r < 8; ++r) c[r] = bias;
        c = wmma_f16(a, bf, c);

        int outbase = (rowbase + hi * 8) * HDIM + ct * 16 + ln;
        if (full) {
#pragma unroll
            for (int r = 0; r < 8; ++r) h[outbase + r * HDIM] = c[r];
        } else {
#pragma unroll
            for (int r = 0; r < 8; ++r)
                if (rowbase + r + 8 * hi < N) h[outbase + r * HDIM] = c[r];
        }
    }
}

// ---------------------------------------------------------------------------
// Edge pass (memory-bound hot loop): recompute e = edge_attr@edge_W + edge_b,
// m = relu(h[src] + e), scatter-add to agg[dst]. 16 threads/edge, 4 ch each.
// ---------------------------------------------------------------------------
__global__ void k_edge_scatter(const float* __restrict__ ea,
                               const int* __restrict__ ei,
                               const float* __restrict__ eW,
                               const float* __restrict__ eb,
                               const float* __restrict__ h,
                               float* __restrict__ agg,
                               int E) {
    __shared__ float sW[16 * HDIM];
    __shared__ float sb[HDIM];
    int t = threadIdx.x;
    for (int i = t; i < 16 * HDIM; i += blockDim.x) sW[i] = eW[i];
    if (t < HDIM) sb[t] = eb[t];
    __syncthreads();

    int gid = blockIdx.x * blockDim.x + t;          // E*16 = 25.6M < 2^31
    if (gid >= E * 16) return;
    int e  = gid >> 4;
    int cg = gid & 15;
    int col = cg * 4;

    int src = ei[e];
    int dst = ei[E + e];

    // prefetch a future edge's feature line into cache (global_prefetch_b8)
    __builtin_prefetch(&ea[(e + 256) * 16], 0, 0);

    float acc0 = sb[col + 0], acc1 = sb[col + 1];
    float acc2 = sb[col + 2], acc3 = sb[col + 3];
    const float* eap = &ea[e * 16];
#pragma unroll
    for (int k = 0; k < 16; ++k) {
        float v = eap[k];
        const float* wr = &sW[k * HDIM + col];
        acc0 = fmaf(v, wr[0], acc0);
        acc1 = fmaf(v, wr[1], acc1);
        acc2 = fmaf(v, wr[2], acc2);
        acc3 = fmaf(v, wr[3], acc3);
    }
    float4 hs = *(const float4*)&h[src * HDIM + col];
    float m0 = fmaxf(hs.x + acc0, 0.0f);
    float m1 = fmaxf(hs.y + acc1, 0.0f);
    float m2 = fmaxf(hs.z + acc2, 0.0f);
    float m3 = fmaxf(hs.w + acc3, 0.0f);
    float* ap = &agg[dst * HDIM + col];
    atomicAdd(ap + 0, m0);
    atomicAdd(ap + 1, m1);
    atomicAdd(ap + 2, m2);
    atomicAdd(ap + 3, m3);
}

// ---------------------------------------------------------------------------
// GINE MLP (WMMA): z = relu((h+agg)@W1 + b1)@W2 + b2, plus BN partial sums.
// 4 waves/block, one 16-row tile per wave. GEMM1: K=64 (2 WMMA) x 8 col tiles.
// GEMM2: K=128 (4 WMMA) x 4 col tiles.
// All LDS operands pre-swizzled to WMMA fragment order (vector ds loads).
// ---------------------------------------------------------------------------
#define MLP_WAVES 4
__global__ void k_mlp(const float* __restrict__ h,
                      const float* __restrict__ agg,
                      const float* __restrict__ W1,
                      const float* __restrict__ b1,
                      const float* __restrict__ W2,
                      const float* __restrict__ b2,
                      float* __restrict__ z,
                      float* __restrict__ chanSum,
                      float* __restrict__ chanSq,
                      int N) {
    // W1: kt<2, ct<8 -> [(kt*8+ct)][lane][i]
    __shared__ __align__(32) _Float16 sW1[16 * 32 * 16];          // 16 KB
    // W2: kt<4, ct<4 -> [(kt*4+ct)][lane][i]
    __shared__ __align__(32) _Float16 sW2[16 * 32 * 16];          // 16 KB
    __shared__ float sb1[H2];
    __shared__ float sb2[HDIM];
    // intermediate t in GEMM2 A-fragment order: [kt][lane][i], per wave
    __shared__ __align__(32) _Float16 sT[MLP_WAVES][4 * 32 * 16]; // 16 KB

    int t = threadIdx.x;
    for (int idx = t; idx < HDIM * H2; idx += blockDim.x) {
        int k = idx >> 7, n = idx & 127;         // W1[k][n], K=64, NC=128
        int kt = k >> 5, hi = (k >> 4) & 1, i = k & 15;
        int ct = n >> 4, ln = n & 15;
        sW1[((kt * 8 + ct) * 32 + hi * 16 + ln) * 16 + i] = (_Float16)W1[idx];
    }
    for (int idx = t; idx < H2 * HDIM; idx += blockDim.x) {
        int k = idx >> 6, n = idx & 63;          // W2[k][n], K=128, NC=64
        int kt = k >> 5, hi = (k >> 4) & 1, i = k & 15;
        int ct = n >> 4, ln = n & 15;
        sW2[((kt * 4 + ct) * 32 + hi * 16 + ln) * 16 + i] = (_Float16)W2[idx];
    }
    if (t < H2)   sb1[t] = b1[t];
    if (t < HDIM) sb2[t] = b2[t];
    __syncthreads();

    int wave = t >> 5, lane = t & 31, hi = lane >> 4, ln = lane & 15;
    int rowbase = (blockIdx.x * MLP_WAVES + wave) * 16;
    int arow = rowbase + ln;
    if (arow >= N) arow = N - 1;     // clamp (only pollutes rows we mask out)
    bool full = (rowbase + 16 <= N); // wave-uniform

    // A fragments for z_in = h + agg (16x64 -> 2 K-tiles)
    v16h a0, a1;
    const float* hr = &h[arow * HDIM];
    const float* gr = &agg[arow * HDIM];
#pragma unroll
    for (int i = 0; i < 16; ++i) {
        int k = a_kmap(i, hi);
        a0[i] = (_Float16)(hr[k] + gr[k]);
        a1[i] = (_Float16)(hr[32 + k] + gr[32 + k]);
    }

    // GEMM1 + ReLU -> sT (stored directly in GEMM2 A-fragment order)
    _Float16* tT = sT[wave];
#pragma unroll
    for (int ct = 0; ct < 8; ++ct) {
        v16h bf0 = *(const v16h*)&sW1[((0 * 8 + ct) * 32 + lane) * 16];
        v16h bf1 = *(const v16h*)&sW1[((1 * 8 + ct) * 32 + lane) * 16];
        v8f c;
        float bias = sb1[ct * 16 + ln];
#pragma unroll
        for (int r = 0; r < 8; ++r) c[r] = bias;
        c = wmma_f16(a0, bf0, c);
        c = wmma_f16(a1, bf1, c);
        // scatter into A-fragment slots: col -> (kt2, hi_c, i); row -> ln_c
        int col = ct * 16 + ln;
        int kt2 = col >> 5;
        int rem = col & 31;
        int hi_c = (rem >> 4) & 1;
        int i2 = rem & 15;
#pragma unroll
        for (int r = 0; r < 8; ++r) {
            int ln_c = r + 8 * hi;               // producer row
            tT[(kt2 * 32 + hi_c * 16 + ln_c) * 16 + i2] =
                (_Float16)fmaxf(c[r], 0.0f);
        }
    }
    __syncthreads();

    // A fragments of t: contiguous vector loads
    v16h a2[4];
#pragma unroll
    for (int kt = 0; kt < 4; ++kt)
        a2[kt] = *(const v16h*)&tT[(kt * 32 + lane) * 16];

    // GEMM2 -> z, plus per-channel sum / sumsq for BatchNorm
#pragma unroll
    for (int ct = 0; ct < 4; ++ct) {
        v8f c;
        float bias = sb2[ct * 16 + ln];
#pragma unroll
        for (int r = 0; r < 8; ++r) c[r] = bias;
#pragma unroll
        for (int kt = 0; kt < 4; ++kt) {
            v16h bf = *(const v16h*)&sW2[((kt * 4 + ct) * 32 + lane) * 16];
            c = wmma_f16(a2[kt], bf, c);
        }
        float psum = 0.0f, psq = 0.0f;
        int outbase = (rowbase + hi * 8) * HDIM + ct * 16 + ln;
        if (full) {
#pragma unroll
            for (int r = 0; r < 8; ++r) {
                float v = c[r];
                z[outbase + r * HDIM] = v;
                psum += v;
                psq  += v * v;
            }
        } else {
#pragma unroll
            for (int r = 0; r < 8; ++r) {
                if (rowbase + r + 8 * hi < N) {
                    float v = c[r];
                    z[outbase + r * HDIM] = v;
                    psum += v;
                    psq  += v * v;
                }
            }
        }
        // lanes n and n+16 hold the same output column (rows 0-7 / 8-15)
        psum += __shfl_down(psum, 16);
        psq  += __shfl_down(psq, 16);
        if (hi == 0) {
            atomicAdd(&chanSum[ct * 16 + ln], psum);
            atomicAdd(&chanSq[ct * 16 + ln], psq);
        }
    }
}

// ---------------------------------------------------------------------------
// BN stat finalize: scale = gamma*rsqrt(var+eps), shift = beta - mu*scale
// ---------------------------------------------------------------------------
__global__ void k_bn_stats(const float* __restrict__ chanSum,
                           const float* __restrict__ chanSq,
                           const float* __restrict__ gamma,
                           const float* __restrict__ beta,
                           float* __restrict__ scale,
                           float* __restrict__ shift, int N) {
    int c = threadIdx.x;
    if (c < HDIM) {
        float inv = 1.0f / (float)N;
        float mu  = chanSum[c] * inv;
        float var = chanSq[c] * inv - mu * mu;
        float s   = gamma[c] * rsqrtf(var + 1e-5f);
        scale[c] = s;
        shift[c] = beta[c] - mu * s;
    }
}

__global__ void k_bn_apply(const float4* __restrict__ z4,
                           const float* __restrict__ scale,
                           const float* __restrict__ shift,
                           float4* __restrict__ h4, int n4) {
    int i = blockIdx.x * blockDim.x + threadIdx.x;
    if (i < n4) {
        int c = (i * 4) & (HDIM - 1);
        float4 v = z4[i];
        float4 o;
        o.x = fmaxf(fmaf(v.x, scale[c + 0], shift[c + 0]), 0.0f);
        o.y = fmaxf(fmaf(v.y, scale[c + 1], shift[c + 1]), 0.0f);
        o.z = fmaxf(fmaf(v.z, scale[c + 2], shift[c + 2]), 0.0f);
        o.w = fmaxf(fmaf(v.w, scale[c + 3], shift[c + 3]), 0.0f);
        h4[i] = o;
    }
}

// ---------------------------------------------------------------------------
// Global mean pool (sum + counts via atomics), then linear head
// ---------------------------------------------------------------------------
__global__ void k_pool(const float* __restrict__ h,
                       const int* __restrict__ batch,
                       float* __restrict__ gsum,
                       float* __restrict__ gcnt, int N) {
    int gid = blockIdx.x * blockDim.x + threadIdx.x;
    if (gid >= N * 16) return;
    int node = gid >> 4;
    int cg = gid & 15;
    int b = batch[node];
    float4 v = *(const float4*)&h[node * HDIM + cg * 4];
    float* gp = &gsum[b * HDIM + cg * 4];
    atomicAdd(gp + 0, v.x);
    atomicAdd(gp + 1, v.y);
    atomicAdd(gp + 2, v.z);
    atomicAdd(gp + 3, v.w);
    if (cg == 0) atomicAdd(&gcnt[b], 1.0f);
}

__global__ void k_final(const float* __restrict__ gsum,
                        const float* __restrict__ gcnt,
                        const float* __restrict__ lW,
                        const float* __restrict__ lb,
                        float* __restrict__ out, int G) {
    int g = threadIdx.x;
    if (g < G) {
        float inv = 1.0f / fmaxf(gcnt[g], 1.0f);
        float acc = lb[0];
        for (int c = 0; c < HDIM; ++c)
            acc = fmaf(gsum[g * HDIM + c] * inv, lW[c], acc);
        out[g] = acc;
    }
}

// ---------------------------------------------------------------------------
extern "C" void kernel_launch(void* const* d_in, const int* in_sizes, int n_in,
                              void* d_out, int out_size, void* d_ws, size_t ws_size,
                              hipStream_t stream) {
    const float* x      = (const float*)d_in[0];
    const float* eattr  = (const float*)d_in[1];
    const int*   eidx   = (const int*)d_in[2];
    const int*   batch  = (const int*)d_in[3];
    const float* nodeW  = (const float*)d_in[4];
    const float* nodeb  = (const float*)d_in[5];
    const float* edgeW  = (const float*)d_in[6];
    const float* edgeb  = (const float*)d_in[7];
    const float* W1s    = (const float*)d_in[8];
    const float* b1s    = (const float*)d_in[9];
    const float* W2s    = (const float*)d_in[10];
    const float* b2s    = (const float*)d_in[11];
    const float* gammas = (const float*)d_in[12];
    const float* betas  = (const float*)d_in[13];
    const float* linW   = (const float*)d_in[14];
    const float* linb   = (const float*)d_in[15];
    float* out = (float*)d_out;

    const int N = in_sizes[0] / 32;        // 50000
    const int E = in_sizes[2] / 2;         // 1.6M
    const int G = 64;
    const int N64 = N * HDIM;              // 3.2M, fits int32

    // workspace carve-out (floats)
    float* ws = (float*)d_ws;
    float* h       = ws;  ws += N64;
    float* z       = ws;  ws += N64;
    float* agg     = ws;  ws += N64;       // chanSum/chanSq must follow agg
    float* chanSum = ws;  ws += HDIM;
    float* chanSq  = ws;  ws += HDIM;
    float* scale   = ws;  ws += HDIM;
    float* shift   = ws;  ws += HDIM;
    float* gsum    = ws;  ws += G * HDIM;
    float* gcnt    = ws;  ws += G;

    const int rowsPerBlock = MLP_WAVES * 16;              // 64
    const int gemmBlocks   = (N + rowsPerBlock - 1) / rowsPerBlock;

    // 1) node embedding (WMMA)
    k_node_embed<<<gemmBlocks, MLP_WAVES * 32, 0, stream>>>(x, nodeW, nodeb, h, N);

    // 2) GINE layers
    for (int l = 0; l < 3; ++l) {
        // zero agg + BN stat accumulators (contiguous in workspace)
        k_zero<<<1024, 256, 0, stream>>>(agg, N64 + 2 * HDIM);
        // edge recompute + gather + relu + scatter-add
        {
            int blocks = (E * 16 + 255) / 256;
            k_edge_scatter<<<blocks, 256, 0, stream>>>(eattr, eidx, edgeW, edgeb,
                                                       h, agg, E);
        }
        // fused MLP (WMMA) + BN partial sums
        k_mlp<<<gemmBlocks, MLP_WAVES * 32, 0, stream>>>(
            h, agg,
            W1s + l * HDIM * H2, b1s + l * H2,
            W2s + l * H2 * HDIM, b2s + l * HDIM,
            z, chanSum, chanSq, N);
        // BN finalize + apply (z -> h)
        k_bn_stats<<<1, 64, 0, stream>>>(chanSum, chanSq,
                                         gammas + l * HDIM, betas + l * HDIM,
                                         scale, shift, N);
        {
            int n4 = N64 / 4;
            int blocks = (n4 + 255) / 256;
            k_bn_apply<<<blocks, 256, 0, stream>>>((const float4*)z, scale, shift,
                                                   (float4*)h, n4);
        }
    }

    // 3) global mean pool + linear head
    k_zero<<<32, 256, 0, stream>>>(gsum, G * HDIM + G);
    {
        int blocks = (N * 16 + 255) / 256;
        k_pool<<<blocks, 256, 0, stream>>>(h, batch, gsum, gcnt, N);
    }
    k_final<<<1, 64, 0, stream>>>(gsum, gcnt, linW, linb, out, G);
}
